// GNNEncoder_53111565582608
// MI455X (gfx1250) — compile-verified, matrix operand order checked by reading, not run on hardware
//
#include <hip/hip_runtime.h>

typedef __attribute__((ext_vector_type(16))) _Float16 v16h;
typedef __attribute__((ext_vector_type(8)))  _Float16 v8h;
typedef __attribute__((ext_vector_type(8)))  float    v8f;
typedef __attribute__((ext_vector_type(4)))  unsigned int v4u;
typedef __attribute__((ext_vector_type(8)))  int v8i;
typedef __attribute__((ext_vector_type(4)))  int v4i;

#define TPB 256

#if __has_builtin(__builtin_amdgcn_tensor_load_to_lds) && \
    __has_builtin(__builtin_amdgcn_s_wait_tensorcnt)
#define USE_TDM 1
#endif

// ---------------------------------------------------------------------------
// WMMA GEMM with TDM/LDS-staged weights.
//   C[M,Nc] = A[M,K] * B[K,Nc] (+ column bias), A/Bt in f16, acc f32.
//   Bt ([Nc,K], B transposed) panel of np columns is DMA'd to LDS once per
//   block (Tensor Data Mover, TENSORcnt), A row-block fragments are loaded
//   once into registers, inner loop is ds_load_b128 + v_wmma only.
//   One wave32 per 16x16 C tile row-block; 8 waves (8 m-tiles) per block.
// ---------------------------------------------------------------------------
template<int KSTEPS>
__global__ void gemm_f16_wmma_lds(const _Float16* __restrict__ A,
                                  const _Float16* __restrict__ Bt,
                                  const float* __restrict__ bias,
                                  float* __restrict__ C,
                                  int M, int Nc, int np)
{
    extern __shared__ _Float16 smem[];          // np * K halfs
    const int K  = KSTEPS * 32;
    const int p0 = blockIdx.y * np;             // first column of this panel

    // ---- stage Bt panel [np rows x K] into LDS ----
#ifdef USE_TDM
    if (threadIdx.x < 32) {                     // wave 0 issues the DMA
        unsigned int lds = (unsigned int)(size_t)(void*)smem;
        unsigned long long ga =
            (unsigned long long)(size_t)(const void*)(Bt + (size_t)p0 * K);
        // D# group 0: count=1, lds_addr, global_addr[56:0], type=2 (image)
        v4u g0;
        g0[0] = 1u;
        g0[1] = lds;
        g0[2] = (unsigned int)ga;
        g0[3] = ((unsigned int)(ga >> 32) & 0x01FFFFFFu) | 0x80000000u;
        // D# group 1: data_size=1 (2B), tensor_dim0=K, tensor_dim1=np,
        //             tile_dim0=K, tile_dim1=np, tensor_dim0_stride=K
        v8i g1;
        g1[0] = (int)(1u << 16);
        g1[1] = (int)(((unsigned)K & 0xFFFFu) << 16);
        g1[2] = (int)((((unsigned)K >> 16) & 0xFFFFu) | (((unsigned)np & 0xFFFFu) << 16));
        g1[3] = (int)((((unsigned)np >> 16) & 0xFFFFu) | (((unsigned)K & 0xFFFFu) << 16));
        g1[4] = (int)((unsigned)np & 0xFFFFu);
        g1[5] = (int)(unsigned)K;
        g1[6] = 0;
        g1[7] = 0;
        v4i gz  = {0, 0, 0, 0};                 // groups 2/3 unused (2-D tile)
        v8i gz8 = {0, 0, 0, 0, 0, 0, 0, 0};     // extra group (6-arg toolchain)
        __builtin_amdgcn_tensor_load_to_lds(g0, g1, gz, gz, gz8, 0);
        __builtin_amdgcn_s_wait_tensorcnt(0);
    }
#else
    {
        const v8h* src = (const v8h*)(Bt + (size_t)p0 * K);
        v8h* dst = (v8h*)smem;
        const int tot = (np * K) / 8;
        for (int i = threadIdx.x; i < tot; i += blockDim.x) dst[i] = src[i];
    }
#endif
    __syncthreads();                            // all waves reach this barrier

    const int wv = threadIdx.x >> 5, lane = threadIdx.x & 31;
    const int mt = blockIdx.x * 8 + wv;
    if (mt >= (M >> 4)) return;                 // wave-uniform, EXEC stays all-1s
    const int r = lane & 15, hi = lane >> 4;
    const int m0 = mt << 4;

    // ---- preload A fragments for this 16-row block (read exactly once) ----
    // 16-bit A fragment layout (ISA 7.12.2): lane m holds K pairs
    // {8*hi..8*hi+7} and {16+8*hi..23+8*hi} -> two contiguous 16B loads.
    const _Float16* arow = A + (size_t)(m0 + r) * K + 8 * hi;
    v16h af[KSTEPS];
#pragma unroll
    for (int ks = 0; ks < KSTEPS; ++ks) {
        v8h a0 = *(const v8h*)(arow + ks * 32);
        v8h a1 = *(const v8h*)(arow + ks * 32 + 16);
#pragma unroll
        for (int i = 0; i < 8; ++i) { af[ks][i] = a0[i]; af[ks][i + 8] = a1[i]; }
    }

    // ---- sweep the panel's n-tiles: LDS reads + WMMA only ----
    const int ntl = np >> 4;
    for (int nt = 0; nt < ntl; ++nt) {
        const _Float16* brow = smem + (size_t)(nt * 16 + r) * K + 8 * hi;
        v8f acc = {};
#pragma unroll
        for (int ks = 0; ks < KSTEPS; ++ks) {
            v8h b0 = *(const v8h*)(brow + ks * 32);
            v8h b1 = *(const v8h*)(brow + ks * 32 + 16);
            v16h b;
#pragma unroll
            for (int i = 0; i < 8; ++i) { b[i] = b0[i]; b[i + 8] = b1[i]; }
            acc = __builtin_amdgcn_wmma_f32_16x16x32_f16(
                      false, af[ks], false, b, (short)0, acc, false, false);
        }
        const int ncol = p0 + nt * 16 + r;
        const float bv = bias ? bias[ncol] : 0.0f;
        // C/D layout: VGPR rr -> row (rr + 8*hi), column = lane&15
#pragma unroll
        for (int rr = 0; rr < 8; ++rr)
            C[(size_t)(m0 + rr + 8 * hi) * Nc + ncol] = acc[rr] + bv;
    }
}

// ---------------------------------------------------------------------------
// Small helpers
// ---------------------------------------------------------------------------
__global__ void fill_f32(float* p, float v, int n)
{
    int i = blockIdx.x * blockDim.x + threadIdx.x;
    if (i < n) p[i] = v;
}

__global__ void to_f16(const float* __restrict__ s, _Float16* __restrict__ d, int n)
{
    int i = blockIdx.x * blockDim.x + threadIdx.x;
    if (i < n) d[i] = (_Float16)s[i];
}

__global__ void addbias_to_f16(const float* __restrict__ s, const float* __restrict__ bias,
                               _Float16* __restrict__ d, int n, int cols)
{
    int i = blockIdx.x * blockDim.x + threadIdx.x;
    if (i < n) d[i] = (_Float16)(s[i] + bias[i % cols]);
}

__global__ void gelu_to_f16(const float* __restrict__ s, _Float16* __restrict__ d, int n)
{
    int i = blockIdx.x * blockDim.x + threadIdx.x;
    if (i < n) {
        float x = s[i];
        d[i] = (_Float16)(0.5f * x * (1.0f + erff(x * 0.70710678118654752f)));
    }
}

// transpose f32 [K,Nc] -> f16 [Nc,K]
__global__ void transp_to_f16(const float* __restrict__ src, _Float16* __restrict__ dst,
                              int K, int Nc)
{
    int i = blockIdx.x * blockDim.x + threadIdx.x;
    if (i >= K * Nc) return;
    int k = i / Nc, n = i - k * Nc;
    dst[(size_t)n * K + k] = (_Float16)src[i];
}

// x @ emb_w + emb_b : [N,16]x[16,128]
__global__ void embed_kernel(const float* __restrict__ x, const float* __restrict__ w,
                             const float* __restrict__ b, float* __restrict__ h, int N)
{
    int i = blockIdx.x * blockDim.x + threadIdx.x;
    if (i >= N * 128) return;
    int n = i >> 7, c = i & 127;
    float acc = b[c];
#pragma unroll
    for (int k = 0; k < 16; ++k) acc += x[n * 16 + k] * w[k * 128 + c];
    h[i] = acc;
}

// self-loop attr: segment-sum of edge_attr and degree over dst
__global__ void loop_attr_acc(const int* __restrict__ ei, const float* __restrict__ ea,
                              float* deg, float* la, int E)
{
    int e = blockIdx.x * blockDim.x + threadIdx.x;
    if (e >= E) return;
    int d = ei[E + e];
    atomicAdd(&deg[d], 1.0f);
    atomicAdd(&la[2 * d + 0], ea[2 * e + 0]);
    atomicAdd(&la[2 * d + 1], ea[2 * e + 1]);
}

__global__ void loop_attr_fin(float* la, const float* deg, int N)
{
    int n = blockIdx.x * blockDim.x + threadIdx.x;
    if (n >= N) return;
    float dg = fmaxf(deg[n], 1.0f);
    la[2 * n + 0] /= dg;
    la[2 * n + 1] /= dg;
}

// head-mean (concat=False layer) + bias: [N,4,128] -> [N,128]
__global__ void head_mean(const float* __restrict__ o, const float* __restrict__ bias,
                          float* __restrict__ t, int N)
{
    int i = blockIdx.x * blockDim.x + threadIdx.x;
    if (i >= N * 128) return;
    int n = i >> 7, c = i & 127;
    const float* p = o + (size_t)n * 512;
    t[i] = 0.25f * (p[c] + p[128 + c] + p[256 + c] + p[384 + c]) + bias[c];
}

// residual + LayerNorm over 128 channels; one wave32 per node
__global__ void residual_ln(const float* __restrict__ x, const float* __restrict__ res,
                            const float* __restrict__ g, const float* __restrict__ b,
                            float* __restrict__ out, int N)
{
    int wid  = blockIdx.x * (blockDim.x >> 5) + (threadIdx.x >> 5);
    int lane = threadIdx.x & 31;
    if (wid >= N) return;
    const float* px = x   + (size_t)wid * 128;
    const float* pr = res + (size_t)wid * 128;
    float v[4], sum = 0.0f;
#pragma unroll
    for (int t = 0; t < 4; ++t) { v[t] = px[t * 32 + lane] + pr[t * 32 + lane]; sum += v[t]; }
    for (int o = 16; o; o >>= 1) sum += __shfl_xor(sum, o, 32);
    float mu = sum * (1.0f / 128.0f);
    float var = 0.0f;
#pragma unroll
    for (int t = 0; t < 4; ++t) { float d = v[t] - mu; var += d * d; }
    for (int o = 16; o; o >>= 1) var += __shfl_xor(var, o, 32);
    float rs = rsqrtf(var * (1.0f / 128.0f) + 1e-5f);
#pragma unroll
    for (int t = 0; t < 4; ++t) {
        int c = t * 32 + lane;
        out[(size_t)wid * 128 + c] = (v[t] - mu) * rs * g[c] + b[c];
    }
}

// ---------------------------------------------------------------------------
// Edge phase (wave32 per edge). xl/xr/out are L2-resident (41 MB each).
// ---------------------------------------------------------------------------
__global__ void edge_score(const int* __restrict__ ei, const float* __restrict__ ea,
                           const float* __restrict__ la,
                           const float* __restrict__ xl, const float* __restrict__ xr,
                           const float* __restrict__ We,   // [2,512]
                           const float* __restrict__ att,  // [4,128]
                           float* __restrict__ score, float* __restrict__ smax,
                           int E, int N)
{
    int wid  = blockIdx.x * (blockDim.x >> 5) + (threadIdx.x >> 5);
    int lane = threadIdx.x & 31;
    int Ep = E + N;
    if (wid >= Ep) return;
    int s, d; float a0, a1;
    if (wid < E) { s = ei[wid]; d = ei[E + wid]; a0 = ea[2 * wid]; a1 = ea[2 * wid + 1]; }
    else         { s = d = wid - E;              a0 = la[2 * s];   a1 = la[2 * s + 1];   }
    const float* pl = xl + (size_t)s * 512;
    const float* pr = xr + (size_t)d * 512;
#pragma unroll
    for (int h = 0; h < 4; ++h) {
        float part = 0.0f;
#pragma unroll
        for (int t = 0; t < 4; ++t) {
            int c = h * 128 + t * 32 + lane;
            float v = pl[c] + pr[c] + a0 * We[c] + a1 * We[512 + c];
            v = v > 0.0f ? v : 0.2f * v;                       // leaky_relu(0.2)
            part += v * att[c];
        }
        for (int o = 16; o; o >>= 1) part += __shfl_xor(part, o, 32);
        if (lane == 0) {
            score[(size_t)wid * 4 + h] = part;
            __hip_atomic_fetch_max(&smax[d * 4 + h], part,
                                   __ATOMIC_RELAXED, __HIP_MEMORY_SCOPE_AGENT);
        }
    }
}

__global__ void edge_softmax_p(const int* __restrict__ ei, float* __restrict__ sc,
                               const float* __restrict__ smax, float* __restrict__ denom,
                               int E, int N)
{
    int i = blockIdx.x * blockDim.x + threadIdx.x;
    int Ep = E + N;
    if (i >= Ep * 4) return;
    int e = i >> 2, h = i & 3;
    int d = (e < E) ? ei[E + e] : (e - E);
    float p = expf(sc[i] - smax[d * 4 + h]);
    sc[i] = p;                                  // score buffer becomes p
    atomicAdd(&denom[d * 4 + h], p);
}

__global__ void edge_scatter(const int* __restrict__ ei, const float* __restrict__ p,
                             const float* __restrict__ denom, const float* __restrict__ xl,
                             float* __restrict__ out, int E, int N)
{
    int wid  = blockIdx.x * (blockDim.x >> 5) + (threadIdx.x >> 5);
    int lane = threadIdx.x & 31;
    int Ep = E + N;
    if (wid >= Ep) return;
    int s, d;
    if (wid < E) { s = ei[wid]; d = ei[E + wid]; }
    else         { s = d = wid - E; }
    const float* pl = xl + (size_t)s * 512;
    float* po = out + (size_t)d * 512;
#pragma unroll
    for (int h = 0; h < 4; ++h) {
        float al = p[(size_t)wid * 4 + h] / (denom[d * 4 + h] + 1e-16f);
#pragma unroll
        for (int t = 0; t < 4; ++t) {
            int j = h * 128 + t * 32 + lane;
            atomicAdd(&po[j], pl[j] * al);
        }
    }
}

// ---------------------------------------------------------------------------
// Host orchestration
// ---------------------------------------------------------------------------
extern "C" void kernel_launch(void* const* d_in, const int* in_sizes, int n_in,
                              void* d_out, int out_size, void* d_ws, size_t ws_size,
                              hipStream_t stream)
{
    const int N = 20000, E = 320000, Ep = E + N;

    const float* x    = (const float*)d_in[0];
    const int*   ei   = (const int*)  d_in[1];
    const float* ea   = (const float*)d_in[2];
    const float* embw = (const float*)d_in[3];
    const float* embb = (const float*)d_in[4];
    const float* linl = (const float*)d_in[5];   // [3,128,512]
    const float* linr = (const float*)d_in[6];   // [3,128,512]
    const float* line = (const float*)d_in[7];   // [3,2,512]
    const float* attw = (const float*)d_in[8];   // [3,4,128]
    const float* cb01 = (const float*)d_in[9];   // [2,512]
    const float* cb2  = (const float*)d_in[10];  // [128]
    const float* pw   = (const float*)d_in[11];  // [2,512,128]
    const float* pb   = (const float*)d_in[12];  // [2,128]
    const float* n1g  = (const float*)d_in[13];
    const float* n1b  = (const float*)d_in[14];
    const float* n2g  = (const float*)d_in[15];
    const float* n2b  = (const float*)d_in[16];
    const float* fw1  = (const float*)d_in[17];  // [3,128,256]
    const float* fb1  = (const float*)d_in[18];  // [3,256]
    const float* fw2  = (const float*)d_in[19];  // [3,256,128]
    const float* fb2  = (const float*)d_in[20];  // [3,128]

    char* base = (char*)d_ws; size_t off = 0;
    auto carve = [&](size_t bytes) -> void* {
        void* p = base + off;
        off += (bytes + 255) & ~(size_t)255;
        return p;
    };
    float*     deg   = (float*)    carve((size_t)N * 4);
    float*     la    = (float*)    carve((size_t)N * 2 * 4);
    float*     h     = (float*)    carve((size_t)N * 128 * 4);
    _Float16*  h16   = (_Float16*) carve((size_t)N * 128 * 2);
    float*     xl    = (float*)    carve((size_t)N * 512 * 4);
    float*     xr    = (float*)    carve((size_t)N * 512 * 4);
    float*     outb  = (float*)    carve((size_t)N * 512 * 4);
    float*     score = (float*)    carve((size_t)Ep * 4 * 4);
    float*     smax  = (float*)    carve((size_t)N * 4 * 4);
    float*     denom = (float*)    carve((size_t)N * 4 * 4);
    float*     t128  = (float*)    carve((size_t)N * 128 * 4);
    float*     f256  = (float*)    carve((size_t)N * 256 * 4);
    _Float16*  f256h = (_Float16*) carve((size_t)N * 256 * 2);
    _Float16*  o512h = (_Float16*) carve((size_t)N * 512 * 2);
    _Float16*  wlt   = (_Float16*) carve((size_t)512 * 128 * 2);
    _Float16*  wrt   = (_Float16*) carve((size_t)512 * 128 * 2);
    _Float16*  prjt  = (_Float16*) carve((size_t)128 * 512 * 2);
    _Float16*  f1t   = (_Float16*) carve((size_t)256 * 128 * 2);
    _Float16*  f2t   = (_Float16*) carve((size_t)128 * 256 * 2);

    auto gemm = [&](const _Float16* A, const _Float16* Bt, const float* bias,
                    float* C, int M, int Nc, int K) {
        int np = Nc;                            // panel width: LDS panel <= 64 KB
        while ((size_t)np * K * 2 > 65536) np >>= 1;
        dim3 grid((M / 16 + 7) / 8, Nc / np);
        size_t sh = (size_t)np * K * 2;
        switch (K / 32) {
            case 4:  gemm_f16_wmma_lds<4><<<grid, TPB, sh, stream>>>(A, Bt, bias, C, M, Nc, np); break;
            case 8:  gemm_f16_wmma_lds<8><<<grid, TPB, sh, stream>>>(A, Bt, bias, C, M, Nc, np); break;
            default: gemm_f16_wmma_lds<16><<<grid, TPB, sh, stream>>>(A, Bt, bias, C, M, Nc, np); break;
        }
    };
    auto gb = [](int n) { return (n + TPB - 1) / TPB; };
    const int ewb = (Ep + 7) / 8;   // wave-per-edge blocks

    // ---- self-loop attrs (fill_value='mean') ----
    (void)hipMemsetAsync(deg, 0, (size_t)N * 4, stream);
    (void)hipMemsetAsync(la,  0, (size_t)N * 2 * 4, stream);
    loop_attr_acc<<<gb(E), TPB, 0, stream>>>(ei, ea, deg, la, E);
    loop_attr_fin<<<gb(N), TPB, 0, stream>>>(la, deg, N);

    // ---- embedding ----
    embed_kernel<<<gb(N * 128), TPB, 0, stream>>>(x, embw, embb, h, N);

    for (int i = 0; i < 3; ++i) {
        // per-layer weight prep: transpose + f16
        transp_to_f16<<<gb(128 * 512), TPB, 0, stream>>>(linl + (size_t)i * 128 * 512, wlt, 128, 512);
        transp_to_f16<<<gb(128 * 512), TPB, 0, stream>>>(linr + (size_t)i * 128 * 512, wrt, 128, 512);
        if (i < 2)
            transp_to_f16<<<gb(512 * 128), TPB, 0, stream>>>(pw + (size_t)i * 512 * 128, prjt, 512, 128);
        transp_to_f16<<<gb(128 * 256), TPB, 0, stream>>>(fw1 + (size_t)i * 128 * 256, f1t, 128, 256);
        transp_to_f16<<<gb(256 * 128), TPB, 0, stream>>>(fw2 + (size_t)i * 256 * 128, f2t, 256, 128);

        // xl = h @ Wl, xr = h @ Wr   (WMMA, TDM-staged weights)
        to_f16<<<gb(N * 128), TPB, 0, stream>>>(h, h16, N * 128);
        gemm(h16, wlt, nullptr, xl, N, 512, 128);
        gemm(h16, wrt, nullptr, xr, N, 512, 128);

        // segment softmax attention + scatter
        fill_f32<<<gb(N * 4), TPB, 0, stream>>>(smax, -3.0e38f, N * 4);
        (void)hipMemsetAsync(denom, 0, (size_t)N * 4 * 4, stream);
        (void)hipMemsetAsync(outb,  0, (size_t)N * 512 * 4, stream);
        edge_score<<<ewb, TPB, 0, stream>>>(ei, ea, la, xl, xr,
                                            line + (size_t)i * 1024,
                                            attw + (size_t)i * 512,
                                            score, smax, E, N);
        edge_softmax_p<<<gb(Ep * 4), TPB, 0, stream>>>(ei, score, smax, denom, E, N);
        edge_scatter<<<ewb, TPB, 0, stream>>>(ei, score, denom, xl, outb, E, N);

        // head combine (+proj for concat layers), residual + LN1
        if (i < 2) {
            addbias_to_f16<<<gb(N * 512), TPB, 0, stream>>>(outb, cb01 + (size_t)i * 512,
                                                            o512h, N * 512, 512);
            gemm(o512h, prjt, pb + (size_t)i * 128, t128, N, 128, 512);
        } else {
            head_mean<<<gb(N * 128), TPB, 0, stream>>>(outb, cb2, t128, N);
        }
        residual_ln<<<(N + 7) / 8, TPB, 0, stream>>>(t128, h,
                                                     n1g + (size_t)i * 128, n1b + (size_t)i * 128,
                                                     h, N);

        // FFN: gelu(h@W1+b1)@W2+b2, residual + LN2
        to_f16<<<gb(N * 128), TPB, 0, stream>>>(h, h16, N * 128);
        gemm(h16, f1t, fb1 + (size_t)i * 256, f256, N, 256, 128);
        gelu_to_f16<<<gb(N * 256), TPB, 0, stream>>>(f256, f256h, N * 256);
        gemm(f256h, f2t, fb2 + (size_t)i * 128, t128, N, 128, 256);
        float* lnout = (i == 2) ? (float*)d_out : h;
        residual_ln<<<(N + 7) / 8, TPB, 0, stream>>>(t128, h,
                                                     n2g + (size_t)i * 128, n2b + (size_t)i * 128,
                                                     lnout, N);
    }
}
